// MultiBranchDH_SFNN_86474871538261
// MI455X (gfx1250) — compile-verified, matrix operand order checked by reading, not run on hardware
//
#include <hip/hip_runtime.h>

// ---------------- problem constants ----------------
#define BB   128
#define TT   250
#define II   700
#define KPAD 704            // I padded to multiple of 32 for WMMA K-steps
#define HH   512
#define BRR  4
#define OO   20
#define NN   (HH*BRR)       // 2048
#define MM   (BB*TT)        // 32000

typedef __bf16 bf16;
typedef __attribute__((ext_vector_type(16))) __bf16 v16bf;
typedef __attribute__((ext_vector_type(8)))  __bf16 v8bf;
typedef __attribute__((ext_vector_type(8)))  float   v8f;

__device__ __forceinline__ bf16 f2bf(float f) {
    // round-to-nearest-even fp32 -> bf16
    unsigned u = __builtin_bit_cast(unsigned, f);
    u = (u + 0x7FFFu + ((u >> 16) & 1u)) >> 16;
    unsigned short s = (unsigned short)u;
    return __builtin_bit_cast(bf16, s);
}

__device__ __forceinline__ float sigm(float x) { return 1.0f / (1.0f + expf(-x)); }

// ---------------- kernel 1: pack X and Wm=W1*mask to padded bf16 ----------------
__global__ __launch_bounds__(256) void pack_kernel(
    const float* __restrict__ X, const float* __restrict__ W1,
    const unsigned char* __restrict__ mask,
    bf16* __restrict__ Xb, bf16* __restrict__ Wb)
{
    const size_t totalX = (size_t)MM * KPAD;
    const size_t totalW = (size_t)NN * KPAD;
    size_t i = (size_t)blockIdx.x * blockDim.x + threadIdx.x;
    size_t stride = (size_t)gridDim.x * blockDim.x;
    for (; i < totalX + totalW; i += stride) {
        if (i < totalX) {
            size_t row = i / KPAD, col = i % KPAD;
            float v = (col < II) ? X[row * II + col] : 0.0f;
            Xb[i] = f2bf(v);
        } else {
            size_t j = i - totalX;
            size_t row = j / KPAD, col = j % KPAD;
            float v = 0.0f;
            if (col < II) {
                size_t src = row * II + col;
                v = W1[src] * (mask[src] ? 1.0f : 0.0f);
            }
            Wb[j] = f2bf(v);
        }
    }
}

// ---------------- kernel 2: K = Xb @ Wb^T + b1, via bf16 WMMA ----------------
// One wave computes a 32(M) x 64(N) tile: 2x4 accumulators, double-buffered
// fragments over K so loads for step k+32 overlap the WMMAs of step k.
// Result written into the d_trace region of d_out (pre-scan "k" values).
union Frag { v16bf v; v8bf h[2]; };

__device__ __forceinline__ void load_frags(
    Frag a[2], Frag b[4],
    const bf16* __restrict__ aBase, const bf16* __restrict__ bBase, int k)
{
#pragma unroll
    for (int i = 0; i < 2; ++i) {
        const bf16* ap = aBase + (size_t)i * 16 * KPAD + k;
        // 16-bit A layout: lanes 0-15 hold K k+0..7 / k+16..23; lanes 16-31 k+8..15 / k+24..31
        a[i].h[0] = *(const v8bf*)(ap);
        a[i].h[1] = *(const v8bf*)(ap + 16);
    }
#pragma unroll
    for (int j = 0; j < 4; ++j) {
        const bf16* bp = bBase + (size_t)j * 16 * KPAD + k;
        b[j].h[0] = *(const v8bf*)(bp);
        b[j].h[1] = *(const v8bf*)(bp + 16);
    }
}

__device__ __forceinline__ void do_wmma(v8f acc[2][4], Frag a[2], Frag b[4])
{
#pragma unroll
    for (int i = 0; i < 2; ++i)
#pragma unroll
        for (int j = 0; j < 4; ++j)
            acc[i][j] = __builtin_amdgcn_wmma_f32_16x16x32_bf16(
                /*neg_a=*/false, a[i].v, /*neg_b=*/false, b[j].v,
                /*c_mod=*/(short)0, acc[i][j], /*reuse_a=*/false, /*reuse_b=*/false);
}

__global__ __launch_bounds__(256) void gemm_wmma_kernel(
    const bf16* __restrict__ Xb, const bf16* __restrict__ Wb,
    const float* __restrict__ b1, float* __restrict__ Kout)
{
    const int nGroups = NN / 64;                       // 32
    int wid   = blockIdx.x * (blockDim.x >> 5) + (threadIdx.x >> 5);
    int mTile = wid / nGroups;                         // [0, 1000)
    int nGrp  = wid % nGroups;
    if (mTile >= MM / 32) return;

    const int m0   = mTile * 32;
    const int n0   = nGrp * 64;
    const int lane = threadIdx.x & 31;
    const int lrow = lane & 15;                        // M (for A) / N (for B) within tile
    const int ksel = (lane >> 4) * 8;                  // K sub-offset per half-wave

    v8f acc[2][4] = {};

    const bf16* aBase = Xb + (size_t)(m0 + lrow) * KPAD + ksel;
    const bf16* bBase = Wb + (size_t)(n0 + lrow) * KPAD + ksel;

    Frag fa[2][2], fb[2][4];                           // ping-pong fragment buffers
    load_frags(fa[0], fb[0], aBase, bBase, 0);

    for (int k = 0; k < KPAD; k += 64) {
        // stream ahead into L2 (global_prefetch_b8)
        __builtin_prefetch(aBase + k + 128, 0, 1);
        __builtin_prefetch(bBase + k + 128, 0, 1);

        load_frags(fa[1], fb[1], aBase, bBase, k + 32);
        do_wmma(acc, fa[0], fb[0]);
        if (k + 64 < KPAD)
            load_frags(fa[0], fb[0], aBase, bBase, k + 64);
        do_wmma(acc, fa[1], fb[1]);
    }

    // Epilogue: D element (M = r + (lane>=16)*8, N = lane%16) in acc VGPR r. Fuse +b1.
#pragma unroll
    for (int i = 0; i < 2; ++i) {
        const int mrow = m0 + i * 16 + (lane >> 4) * 8;
#pragma unroll
        for (int j = 0; j < 4; ++j) {
            int col = n0 + j * 16 + lrow;
            float bias = b1[col];
            float* outp = Kout + (size_t)mrow * NN + col;
#pragma unroll
            for (int r = 0; r < 8; ++r)
                outp[(size_t)r * NN] = acc[i][j][r] + bias;
        }
    }
}

// ---------------- kernel 3: dendritic + LIF scan over T (in-place on d_trace) ----------------
__global__ __launch_bounds__(256) void scan_hidden_kernel(
    const float* __restrict__ tau_m, const float* __restrict__ tau_n,
    float* __restrict__ mem_tr, float* __restrict__ spk_tr,
    float* __restrict__ dtr /* holds k on entry, d on exit */)
{
    int idx = blockIdx.x * blockDim.x + threadIdx.x;   // B*H = 65536
    if (idx >= BB * HH) return;
    int b = idx / HH, h = idx % HH;

    float alpha = sigm(tau_m[h]);
    float be0 = sigm(tau_n[h * BRR + 0]);
    float be1 = sigm(tau_n[h * BRR + 1]);
    float be2 = sigm(tau_n[h * BRR + 2]);
    float be3 = sigm(tau_n[h * BRR + 3]);

    float4 d = make_float4(0.f, 0.f, 0.f, 0.f);
    float mem = 0.f, spk = 0.f;

    float* kp = dtr + (size_t)b * TT * NN + (size_t)h * BRR;
    float* mp = mem_tr + (size_t)b * TT * HH + h;
    float* sp = spk_tr + (size_t)b * TT * HH + h;

    for (int t = 0; t < TT; ++t) {
        __builtin_prefetch(kp + 4 * NN, 0, 1);         // stream 4 timesteps ahead
        float4 k = *(const float4*)kp;
        d.x = be0 * d.x + (1.f - be0) * k.x;
        d.y = be1 * d.y + (1.f - be1) * k.y;
        d.z = be2 * d.z + (1.f - be2) * k.z;
        d.w = be3 * d.w + (1.f - be3) * k.w;
        *(float4*)kp = d;
        float l = (d.x + d.y) + (d.z + d.w);
        mem = alpha * mem - spk + (1.f - alpha) * l;   // V_TH = 1
        spk = (mem - 1.0f > 0.0f) ? 1.0f : 0.0f;
        *mp = mem;
        *sp = spk;
        kp += NN; mp += HH; sp += HH;
    }
}

// ---------------- kernel 4: z = spk @ W2^T + b2 into output region ----------------
__global__ __launch_bounds__(128) void readout_kernel(
    const float* __restrict__ spk_tr, const float* __restrict__ W2,
    const float* __restrict__ b2, float* __restrict__ out)
{
    __shared__ float s[HH];
    int bt = blockIdx.x;                               // 32000
    const float* row = spk_tr + (size_t)bt * HH;
    for (int i = threadIdx.x; i < HH; i += blockDim.x) s[i] = row[i];
    __syncthreads();
    if (threadIdx.x < OO) {
        int o = threadIdx.x;
        float acc = b2[o];
        const float* w = W2 + (size_t)o * HH;
        for (int j = 0; j < HH; ++j) acc = fmaf(s[j], w[j], acc);
        out[(size_t)bt * OO + o] = acc;
    }
}

// ---------------- kernel 5: leaky readout scan over T (in-place on output) ----------------
__global__ __launch_bounds__(256) void scan_out_kernel(
    const float* __restrict__ tau_m_out, float* __restrict__ out)
{
    int idx = blockIdx.x * blockDim.x + threadIdx.x;   // B*O = 2560
    if (idx >= BB * OO) return;
    int b = idx / OO, o = idx % OO;
    float ao = sigm(tau_m_out[o]);
    float m = 0.f;
    float* p = out + (size_t)b * TT * OO + o;
    for (int t = 0; t < TT; ++t) {
        float z = *p;
        m = ao * m + (1.f - ao) * z;
        *p = m;
        p += OO;
    }
}

extern "C" void kernel_launch(void* const* d_in, const int* in_sizes, int n_in,
                              void* d_out, int out_size, void* d_ws, size_t ws_size,
                              hipStream_t stream) {
    const float*         X         = (const float*)d_in[0];
    const float*         W1        = (const float*)d_in[1];
    const float*         b1        = (const float*)d_in[2];
    const float*         tau_m     = (const float*)d_in[3];
    const float*         tau_n     = (const float*)d_in[4];
    const float*         W2        = (const float*)d_in[5];
    const float*         b2        = (const float*)d_in[6];
    const float*         tau_m_out = (const float*)d_in[7];
    const unsigned char* mask      = (const unsigned char*)d_in[8]; // bool mask

    // d_out layout: output | mem_trace | spike_trace | d_trace (flat, return order)
    float* out    = (float*)d_out;
    float* mem_tr = out    + (size_t)BB * TT * OO;     // +640,000
    float* spk_tr = mem_tr + (size_t)BB * TT * HH;     // +16,384,000
    float* dtr    = spk_tr + (size_t)BB * TT * HH;     // +16,384,000 -> 65,536,000 elems

    // workspace: bf16 X (padded) then bf16 Wm (padded) : ~48 MB
    bf16* Xb = (bf16*)d_ws;
    bf16* Wb = Xb + (size_t)MM * KPAD;

    // 1) pack / mask / convert
    {
        size_t total = (size_t)MM * KPAD + (size_t)NN * KPAD;
        int blocks = (int)((total + 255) / 256);
        pack_kernel<<<blocks, 256, 0, stream>>>(X, W1, mask, Xb, Wb);
    }
    // 2) big GEMM: K -> d_trace region. 1000 mTiles * 32 nGroups = 32000 waves = 4000 blocks
    gemm_wmma_kernel<<<4000, 256, 0, stream>>>(Xb, Wb, b1, dtr);
    // 3) hidden scans (d in-place, mem/spike traces)
    scan_hidden_kernel<<<(BB * HH + 255) / 256, 256, 0, stream>>>(tau_m, tau_n, mem_tr, spk_tr, dtr);
    // 4) readout GEMM into output region
    readout_kernel<<<MM, 128, 0, stream>>>(spk_tr, W2, b2, out);
    // 5) leaky output scan in-place
    scan_out_kernel<<<(BB * OO + 255) / 256, 256, 0, stream>>>(tau_m_out, out);
}
